// Attention_68126771249186
// MI455X (gfx1250) — matrix operand order repaired and checked by reading, not compiled, not run on hardware
//
#include <hip/hip_runtime.h>
#include <hip/hip_bf16.h>

typedef _Float16 h16 __attribute__((ext_vector_type(16)));
typedef _Float16 h8  __attribute__((ext_vector_type(8)));
typedef float    f8  __attribute__((ext_vector_type(8)));

union H16u { h16 v; h8 h[2]; };

// Assemble a WMMA A fragment from two contiguous 16B chunks at p and p+16B.
__device__ __forceinline__ h16 ld_a16(const _Float16* p) {
    H16u u;
    u.h[0] = *(const h8*)p;
    u.h[1] = *(const h8*)(p + 8);
    return u.v;
}

__device__ __forceinline__ f8 wmma_f16(h16 a, h16 b, f8 c) {
    return __builtin_amdgcn_wmma_f32_16x16x32_f16(false, a, false, b, (short)0, c, false, false);
}

// ---------------- Kernel 1: pixel-unshuffle + f32->f16 tokens [8192][4096] ----------------
__global__ void pack_tokens_k(const float* __restrict__ x, _Float16* __restrict__ tok) {
    int i = blockIdx.x * 256 + threadIdx.x;     // 8*64*256*256 = 33,554,432 exact
    int w0 = i & 255, h0 = (i >> 8) & 255, c0 = (i >> 16) & 63, b = i >> 22;
    int c = (c0 << 6) + ((h0 & 7) << 3) + (w0 & 7);
    int n = ((h0 >> 3) << 5) + (w0 >> 3);
    tok[(size_t)(((b << 10) + n) << 12) + c] = (_Float16)x[i];
}

// ---------------- Kernel 2/3: pack weight into per-lane WMMA-B layout ----------------
// P layout: [kt][nt][lane(32)][16 halves], tile = 512 elements
__global__ void pack_b_k(const float* __restrict__ W, _Float16* __restrict__ P, int Ncols, int total) {
    int i = blockIdx.x * 256 + threadIdx.x;
    if (i >= total) return;
    int j = i & 15, lane = (i >> 4) & 31;
    int t = i >> 9;
    int NT = Ncols >> 4;
    int nt = t % NT, kt = t / NT;
    int K = kt * 32 + ((lane >> 4) << 4) + j;
    int N = (nt << 4) + (lane & 15);
    P[i] = (_Float16)W[(size_t)K * Ncols + N];
}

// ---------------- Kernel 4: QKV GEMM  [8192 x 4096] x [4096 x 768] ----------------
// 1 wave/block, 32x64 tile, explicit ping-pong double buffering (no rotation moves).
__global__ void __launch_bounds__(32)
gemm1_k(const _Float16* __restrict__ tok, const _Float16* __restrict__ bp,
        _Float16* __restrict__ q, _Float16* __restrict__ kk, _Float16* __restrict__ vT) {
    int lane = threadIdx.x;
    int r = lane & 15, sub = lane >> 4;
    int m0 = blockIdx.x << 5;
    int ntb = blockIdx.y << 2;                  // base 16-col tile (48 total)

    const _Float16* pa0 = tok + (size_t)(m0 + r) * 4096 + sub * 16;
    const _Float16* pa1 = pa0 + 16 * 4096;
    const _Float16* pb  = bp + (size_t)((ntb << 5) + lane) * 16;

    f8 acc[2][4] = {};

    // buffer 0 <- kt 0
    h16 a00 = ld_a16(pa0), a01 = ld_a16(pa1);
    h16 b00 = *(const h16*)(pb);
    h16 b01 = *(const h16*)(pb + 512);
    h16 b02 = *(const h16*)(pb + 1024);
    h16 b03 = *(const h16*)(pb + 1536);
    // buffer 1 <- kt 1
    pa0 += 32; pa1 += 32; pb += 48 * 512;
    h16 a10 = ld_a16(pa0), a11 = ld_a16(pa1);
    h16 b10 = *(const h16*)(pb);
    h16 b11 = *(const h16*)(pb + 512);
    h16 b12 = *(const h16*)(pb + 1024);
    h16 b13 = *(const h16*)(pb + 1536);

#pragma unroll 1
    for (int i = 0; i < 63; ++i) {
        // compute kt = 2i (buf0)
        acc[0][0] = wmma_f16(a00, b00, acc[0][0]);
        acc[1][0] = wmma_f16(a01, b00, acc[1][0]);
        acc[0][1] = wmma_f16(a00, b01, acc[0][1]);
        acc[1][1] = wmma_f16(a01, b01, acc[1][1]);
        acc[0][2] = wmma_f16(a00, b02, acc[0][2]);
        acc[1][2] = wmma_f16(a01, b02, acc[1][2]);
        acc[0][3] = wmma_f16(a00, b03, acc[0][3]);
        acc[1][3] = wmma_f16(a01, b03, acc[1][3]);
        // load buf0 <- kt = 2i+2
        pa0 += 32; pa1 += 32; pb += 48 * 512;
        a00 = ld_a16(pa0); a01 = ld_a16(pa1);
        b00 = *(const h16*)(pb);
        b01 = *(const h16*)(pb + 512);
        b02 = *(const h16*)(pb + 1024);
        b03 = *(const h16*)(pb + 1536);
        // compute kt = 2i+1 (buf1)
        acc[0][0] = wmma_f16(a10, b10, acc[0][0]);
        acc[1][0] = wmma_f16(a11, b10, acc[1][0]);
        acc[0][1] = wmma_f16(a10, b11, acc[0][1]);
        acc[1][1] = wmma_f16(a11, b11, acc[1][1]);
        acc[0][2] = wmma_f16(a10, b12, acc[0][2]);
        acc[1][2] = wmma_f16(a11, b12, acc[1][2]);
        acc[0][3] = wmma_f16(a10, b13, acc[0][3]);
        acc[1][3] = wmma_f16(a11, b13, acc[1][3]);
        // load buf1 <- kt = 2i+3
        pa0 += 32; pa1 += 32; pb += 48 * 512;
        a10 = ld_a16(pa0); a11 = ld_a16(pa1);
        b10 = *(const h16*)(pb);
        b11 = *(const h16*)(pb + 512);
        b12 = *(const h16*)(pb + 1024);
        b13 = *(const h16*)(pb + 1536);
    }
    // compute kt = 126 (buf0) and kt = 127 (buf1)
    acc[0][0] = wmma_f16(a00, b00, acc[0][0]);
    acc[1][0] = wmma_f16(a01, b00, acc[1][0]);
    acc[0][1] = wmma_f16(a00, b01, acc[0][1]);
    acc[1][1] = wmma_f16(a01, b01, acc[1][1]);
    acc[0][2] = wmma_f16(a00, b02, acc[0][2]);
    acc[1][2] = wmma_f16(a01, b02, acc[1][2]);
    acc[0][3] = wmma_f16(a00, b03, acc[0][3]);
    acc[1][3] = wmma_f16(a01, b03, acc[1][3]);
    acc[0][0] = wmma_f16(a10, b10, acc[0][0]);
    acc[1][0] = wmma_f16(a11, b10, acc[1][0]);
    acc[0][1] = wmma_f16(a10, b11, acc[0][1]);
    acc[1][1] = wmma_f16(a11, b11, acc[1][1]);
    acc[0][2] = wmma_f16(a10, b12, acc[0][2]);
    acc[1][2] = wmma_f16(a11, b12, acc[1][2]);
    acc[0][3] = wmma_f16(a10, b13, acc[0][3]);
    acc[1][3] = wmma_f16(a11, b13, acc[1][3]);

    int laneN = lane & 15, hiM = sub << 3;
#pragma unroll
    for (int mt = 0; mt < 2; ++mt)
#pragma unroll
        for (int nt = 0; nt < 4; ++nt) {
            int col = ((ntb + nt) << 4) + laneN;
            int part = col >> 8, h = (col >> 6) & 3, dd = col & 63;
#pragma unroll
            for (int rr = 0; rr < 8; ++rr) {
                int m = m0 + mt * 16 + hiM + rr;
                int b = m >> 10, n = m & 1023;
                int bh = (b << 2) + h;
                float val = acc[mt][nt][rr];
                if (part == 0)      q [(size_t)(((bh << 10) + n) << 6) + dd] = (_Float16)(val * 0.125f);
                else if (part == 1) kk[(size_t)(((bh << 10) + n) << 6) + dd] = (_Float16)val;
                else                vT[(size_t)((bh << 6) + dd) * 1024 + n]  = (_Float16)val;
            }
        }
}

// ---------------- Kernel 5: attention per (b, h, 32-row tile) ----------------
__global__ void __launch_bounds__(128)
attn_k(const _Float16* __restrict__ q, const _Float16* __restrict__ kk,
       const _Float16* __restrict__ vT, _Float16* __restrict__ obuf) {
    extern __shared__ char smem[];
    float*    dots = (float*)smem;                         // [32][1024] f32 = 128 KB
    _Float16* attn = (_Float16*)(smem + 32 * 1024 * 4);    // [32][1024] f16 =  64 KB
    __shared__ float red [32][4];
    __shared__ float red2[32][4];

    int tid  = threadIdx.x;
    int lane = tid & 31;
    int wave = tid >> 5;
    int mtile = blockIdx.x & 31;
    int h = (blockIdx.x >> 5) & 3;
    int b = blockIdx.x >> 7;
    int bh = (b << 2) + h;
    const _Float16* qb = q  + (size_t)(bh << 10) * 64;
    const _Float16* kb = kk + (size_t)(bh << 10) * 64;
    const _Float16* vb = vT + (size_t)bh * 64 * 1024;
    int m0 = mtile << 5;
    int r = lane & 15, sub = lane >> 4;
    int laneN = r, hiM = sub << 3;

    // ---- phase 1: dots = (q*scale) @ k^T  -> LDS f32 ----
    {
        const _Float16* pq = qb + (size_t)(m0 + r) * 64 + sub * 16;
        h16 afr00 = ld_a16(pq);
        h16 afr01 = ld_a16(pq + 32);
        h16 afr10 = ld_a16(pq + 16 * 64);
        h16 afr11 = ld_a16(pq + 16 * 64 + 32);
        const _Float16* pk = kb + (size_t)((wave << 4) + r) * 64 + sub * 16;
        float* pdA = dots + hiM * 1024 + (wave << 4) + laneN;
        float* pdB = pdA + 16 * 1024;
#pragma unroll 1
        for (int it = 0; it < 16; ++it) {
            f8 acc0 = {}, acc1 = {};
            h16 b0 = *(const h16*)(pk);
            h16 b1 = *(const h16*)(pk + 32);
            acc0 = wmma_f16(afr00, b0, acc0);
            acc0 = wmma_f16(afr01, b1, acc0);
            acc1 = wmma_f16(afr10, b0, acc1);
            acc1 = wmma_f16(afr11, b1, acc1);
#pragma unroll
            for (int rr = 0; rr < 8; ++rr) {
                pdA[rr * 1024] = acc0[rr];
                pdB[rr * 1024] = acc1[rr];
            }
            pk += 4 * 16 * 64;    // next nt (stride 4 tiles of 16 rows)
            pdA += 64; pdB += 64;
        }
    }
    __syncthreads();

    // ---- phase 2: row softmax (4 threads per row, 256 elems each) ----
    {
        int row = tid >> 2, seg = (tid & 3) << 8;
        const float* dr = dots + row * 1024 + seg;
        float mx = -3.0e38f;
        for (int j = 0; j < 256; ++j) mx = fmaxf(mx, dr[j]);
        red[row][tid & 3] = mx;
        __syncthreads();
        float rmax = fmaxf(fmaxf(red[row][0], red[row][1]), fmaxf(red[row][2], red[row][3]));
        float s = 0.f;
        for (int j = 0; j < 256; ++j) s += __expf(dr[j] - rmax);
        red2[row][tid & 3] = s;
        __syncthreads();
        float inv = 1.f / (red2[row][0] + red2[row][1] + red2[row][2] + red2[row][3]);
        _Float16* ar = attn + row * 1024 + seg;
        for (int j = 0; j < 256; ++j) ar[j] = (_Float16)(__expf(dr[j] - rmax) * inv);
    }
    __syncthreads();

    // ---- phase 3: out = attn @ v ; wave w owns d-columns [16w, 16w+16), ping-pong ----
    {
        int nt = wave;
        f8 acc0 = {}, acc1 = {};
        const _Float16* pv = vb + (size_t)((nt << 4) + r) * 1024 + sub * 16;
        const _Float16* pt0 = attn + (size_t)r * 1024 + sub * 16;
        const _Float16* pt1 = pt0 + 16 * 1024;
        // buf0 <- step 0
        h16 v0 = *(const h16*)(pv);
        h16 x00 = ld_a16(pt0);
        h16 x01 = ld_a16(pt1);
        // buf1 <- step 1
        pv += 32; pt0 += 32; pt1 += 32;
        h16 v1 = *(const h16*)(pv);
        h16 x10 = ld_a16(pt0);
        h16 x11 = ld_a16(pt1);
#pragma unroll 1
        for (int i = 0; i < 15; ++i) {
            acc0 = wmma_f16(x00, v0, acc0);
            acc1 = wmma_f16(x01, v0, acc1);
            pv += 32; pt0 += 32; pt1 += 32;
            v0 = *(const h16*)(pv);
            x00 = ld_a16(pt0);
            x01 = ld_a16(pt1);
            acc0 = wmma_f16(x10, v1, acc0);
            acc1 = wmma_f16(x11, v1, acc1);
            pv += 32; pt0 += 32; pt1 += 32;
            v1 = *(const h16*)(pv);
            x10 = ld_a16(pt0);
            x11 = ld_a16(pt1);
        }
        acc0 = wmma_f16(x00, v0, acc0);
        acc1 = wmma_f16(x01, v0, acc1);
        acc0 = wmma_f16(x10, v1, acc0);
        acc1 = wmma_f16(x11, v1, acc1);

        int colBase = (h << 6) + (nt << 4) + laneN;
#pragma unroll
        for (int rr = 0; rr < 8; ++rr) {
            int n0 = m0 + hiM + rr;
            obuf[(size_t)(((b << 10) + n0) << 8) + colBase] = (_Float16)acc0[rr];
            int n1 = m0 + 16 + hiM + rr;
            obuf[(size_t)(((b << 10) + n1) << 8) + colBase] = (_Float16)acc1[rr];
        }
    }
}

// ---------------- Kernel 6: out-proj GEMM + bias + pixel-shuffle scatter ----------------
__global__ void __launch_bounds__(32)
gemm2_k(const _Float16* __restrict__ obuf, const _Float16* __restrict__ bp,
        const float* __restrict__ bias, float* __restrict__ out) {
    int lane = threadIdx.x;
    int r = lane & 15, sub = lane >> 4;
    int m0 = blockIdx.x << 5;
    int ntb = blockIdx.y << 2;                 // base 16-col tile (256 total)

    const _Float16* pa0 = obuf + (size_t)(m0 + r) * 256 + sub * 16;
    const _Float16* pa1 = pa0 + 16 * 256;
    const _Float16* pbB = bp + (size_t)((ntb << 5) + lane) * 16;

    f8 acc[2][4] = {};
#pragma unroll 2
    for (int kt = 0; kt < 8; ++kt) {
        h16 a0 = ld_a16(pa0 + kt * 32);
        h16 a1 = ld_a16(pa1 + kt * 32);
        const _Float16* pb = pbB + (size_t)kt * 256 * 512;
#pragma unroll
        for (int nt = 0; nt < 4; ++nt) {
            h16 bb = *(const h16*)(pb + nt * 512);
            acc[0][nt] = wmma_f16(a0, bb, acc[0][nt]);
            acc[1][nt] = wmma_f16(a1, bb, acc[1][nt]);
        }
    }
    int laneN = r, hiM = sub << 3;
#pragma unroll
    for (int mt = 0; mt < 2; ++mt)
#pragma unroll
        for (int nt = 0; nt < 4; ++nt) {
            int c = ((ntb + nt) << 4) + laneN;
            float bv = bias[c];
            int c0 = c >> 6, r1 = (c >> 3) & 7, r2 = c & 7;
#pragma unroll
            for (int rr = 0; rr < 8; ++rr) {
                int m = m0 + mt * 16 + hiM + rr;
                int b = m >> 10, n = m & 1023;
                int hi = n >> 5, wi = n & 31;
                out[((size_t)(((b << 6) + c0) << 8) + (hi << 3) + r1) * 256 + (wi << 3) + r2] =
                    acc[mt][nt][rr] + bv;
            }
        }
}

extern "C" void kernel_launch(void* const* d_in, const int* in_sizes, int n_in,
                              void* d_out, int out_size, void* d_ws, size_t ws_size,
                              hipStream_t stream) {
    (void)in_sizes; (void)n_in; (void)out_size; (void)ws_size;
    const float* x    = (const float*)d_in[0];
    const float* Wqkv = (const float*)d_in[1];
    const float* Wout = (const float*)d_in[2];
    const float* bout = (const float*)d_in[3];
    float* out = (float*)d_out;

    char* ws = (char*)d_ws;
    _Float16* tok = (_Float16*)ws;  ws += (size_t)8192 * 4096 * 2;    // tokens f16
    _Float16* wqp = (_Float16*)ws;  ws += (size_t)4096 * 768 * 2;     // W_qkv packed
    _Float16* wop = (_Float16*)ws;  ws += (size_t)256 * 4096 * 2;     // W_out packed
    _Float16* qb  = (_Float16*)ws;  ws += (size_t)32 * 1024 * 64 * 2; // q (scaled)
    _Float16* kb  = (_Float16*)ws;  ws += (size_t)32 * 1024 * 64 * 2; // k
    _Float16* vT  = (_Float16*)ws;  ws += (size_t)32 * 1024 * 64 * 2; // v transposed
    _Float16* ob  = (_Float16*)ws;  ws += (size_t)8192 * 256 * 2;     // attn output

    hipLaunchKernelGGL(pack_tokens_k, dim3(131072), dim3(256), 0, stream, x, tok);
    hipLaunchKernelGGL(pack_b_k, dim3((4096 * 768 + 255) / 256), dim3(256), 0, stream,
                       Wqkv, wqp, 768, 4096 * 768);
    hipLaunchKernelGGL(pack_b_k, dim3((256 * 4096 + 255) / 256), dim3(256), 0, stream,
                       Wout, wop, 4096, 256 * 4096);
    hipLaunchKernelGGL(gemm1_k, dim3(256, 12), dim3(32), 0, stream, tok, wqp, qb, kb, vT);
    (void)hipFuncSetAttribute((const void*)attn_k,
                              hipFuncAttributeMaxDynamicSharedMemorySize, 196608);
    hipLaunchKernelGGL(attn_k, dim3(1024), dim3(128), 196608, stream, qb, kb, vT, ob);
    hipLaunchKernelGGL(gemm2_k, dim3(256, 64), dim3(32), 0, stream, ob, wop, bout, out);
}
